// GRUModel_66589172957321
// MI455X (gfx1250) — matrix operand order repaired and checked by reading, not compiled, not run on hardware
//
#include <hip/hip_runtime.h>

typedef __attribute__((ext_vector_type(16))) _Float16 v16h;
typedef __attribute__((ext_vector_type(8)))  _Float16 v8h;
typedef __attribute__((ext_vector_type(8)))  float    v8f;
typedef __attribute__((ext_vector_type(4)))  float    v4f;

#define WMMA16(a,b,c) __builtin_amdgcn_wmma_f32_16x16x32_f16(false,(a),false,(b),(short)0,(c),false,false)

static constexpr int Bb = 2048, Tt = 512, Dd = 32, Hh = 64;

// ---------------- helpers ----------------

// fast sigmoid / tanh using v_rcp_f32 (no IEEE division sequences)
static __device__ __forceinline__ float sig_f(float v) {
    return __builtin_amdgcn_rcpf(1.0f + __expf(-v));
}
static __device__ __forceinline__ float tanh_f(float v) {
    float e = __expf(2.0f * v);
    return 1.0f - 2.0f * __builtin_amdgcn_rcpf(e + 1.0f);   // robust at +/-inf
}
static __device__ __forceinline__ v8f splat8(float s) {
    v8f r;
#pragma unroll
    for (int i = 0; i < 8; ++i) r[i] = s;
    return r;
}

// B-fragment (K x 16, 16-bit): lane ln holds column N=ln; lanes 0-15 K=kbase+0..7 and kbase+16..23,
// lanes 16-31: +8 on K. B[k,n] = W[nbase+n, k] (we compute h @ W^T).
static __device__ __forceinline__ v16h ldB(const _Float16* __restrict__ W, int K,
                                           int nbase, int kbase, int ln, int hl) {
    const _Float16* p = W + (size_t)(nbase + ln) * K + kbase + (hl << 3);
    v8h lo = *(const v8h*)p;
    v8h hi = *(const v8h*)(p + 16);
    return __builtin_shufflevector(lo, hi, 0,1,2,3,4,5,6,7,8,9,10,11,12,13,14,15);
}

// A-fragment (16 x 32, 16-bit) from row-major LDS tile [16 x 64] f16.
// lanes 0-15: M=ln, K kbase+0..7 (v0..3) and kbase+16..23 (v4..7); lanes 16-31: +8 on K.
static __device__ __forceinline__ v16h ldA(const _Float16* base, int kbase, int ln, int hl) {
    const _Float16* p = base + ln * 64 + kbase + (hl << 3);
    v8h lo = *(const v8h*)p;          // ds_load_b128
    v8h hi = *(const v8h*)(p + 16);   // ds_load_b128
    return __builtin_shufflevector(lo, hi, 0,1,2,3,4,5,6,7,8,9,10,11,12,13,14,15);
}

// ---------------- prep: f32 -> f16 weight convert ----------------

__global__ void cvt_f16_kernel(const float* __restrict__ src, _Float16* __restrict__ dst, int n) {
    int i = blockIdx.x * blockDim.x + threadIdx.x;
    if (i < n) dst[i] = (_Float16)src[i];
}

// ---------------- fused 2-layer GRU + head ----------------
// block = 128 threads (4 waves), handles 16 batch rows for all T steps.
// wave w owns hidden columns [16w, 16w+16) => gate N-tiles {w, 4+w, 8+w} of the 12.
// Recurrent weights (Bhh0/Bhh1, 12 frags = 96 VGPRs) live in registers (critical path);
// input-projection weights (9 frags) live in LDS as pre-formatted per-lane fragments.

__global__ __launch_bounds__(128, 1) void gru2_fused_wmma(
    const float* __restrict__ x,
    const _Float16* __restrict__ wih0, const _Float16* __restrict__ whh0,
    const _Float16* __restrict__ wih1, const _Float16* __restrict__ whh1,
    const float* __restrict__ b_ih0, const float* __restrict__ b_hh0,
    const float* __restrict__ b_ih1, const float* __restrict__ b_hh1,
    const float* __restrict__ fc_w,  const float* __restrict__ fc_b,
    float* __restrict__ out)
{
    __shared__ _Float16 lh0[16 * 64];        // layer-0 hidden state tile, f16 row-major
    __shared__ _Float16 lh1[16 * 64];        // layer-1 hidden state tile
    __shared__ _Float16 wfr[9 * 128 * 16];   // 9 pre-formatted weight frags per thread (36 KB)
    __shared__ float    red[16 * 64];        // head reduction

    const int tid = threadIdx.x;
    const int lane = tid & 31;
    const int w  = tid >> 5;     // wave id 0..3
    const int hl = lane >> 4;    // half-wave select
    const int ln = lane & 15;
    const int b0 = blockIdx.x * 16;
    const int c  = w * 16 + ln;  // this lane's hidden column (0..63)

    const int nr = w * 16, nz = 64 + w * 16, nn = 128 + w * 16;

    // ---- register-resident recurrent weight fragments ----
    v16h Bhh0_r0 = ldB(whh0, 64, nr, 0,  ln, hl), Bhh0_r1 = ldB(whh0, 64, nr, 32, ln, hl);
    v16h Bhh0_z0 = ldB(whh0, 64, nz, 0,  ln, hl), Bhh0_z1 = ldB(whh0, 64, nz, 32, ln, hl);
    v16h Bhh0_n0 = ldB(whh0, 64, nn, 0,  ln, hl), Bhh0_n1 = ldB(whh0, 64, nn, 32, ln, hl);
    v16h Bhh1_r0 = ldB(whh1, 64, nr, 0,  ln, hl), Bhh1_r1 = ldB(whh1, 64, nr, 32, ln, hl);
    v16h Bhh1_z0 = ldB(whh1, 64, nz, 0,  ln, hl), Bhh1_z1 = ldB(whh1, 64, nz, 32, ln, hl);
    v16h Bhh1_n0 = ldB(whh1, 64, nn, 0,  ln, hl), Bhh1_n1 = ldB(whh1, 64, nn, 32, ln, hl);

    // ---- input-projection fragments parked in LDS (per-lane pre-formatted) ----
    {
        _Float16* s = &wfr[tid * 16];
        *(v16h*)(s + 0 * 128 * 16) = ldB(wih0, 32, nr, 0,  ln, hl);
        *(v16h*)(s + 1 * 128 * 16) = ldB(wih0, 32, nz, 0,  ln, hl);
        *(v16h*)(s + 2 * 128 * 16) = ldB(wih0, 32, nn, 0,  ln, hl);
        *(v16h*)(s + 3 * 128 * 16) = ldB(wih1, 64, nr, 0,  ln, hl);
        *(v16h*)(s + 4 * 128 * 16) = ldB(wih1, 64, nr, 32, ln, hl);
        *(v16h*)(s + 5 * 128 * 16) = ldB(wih1, 64, nz, 0,  ln, hl);
        *(v16h*)(s + 6 * 128 * 16) = ldB(wih1, 64, nz, 32, ln, hl);
        *(v16h*)(s + 7 * 128 * 16) = ldB(wih1, 64, nn, 0,  ln, hl);
        *(v16h*)(s + 8 * 128 * 16) = ldB(wih1, 64, nn, 32, ln, hl);
    }
    const _Float16* wf = &wfr[tid * 16];

    // ---- per-lane gate biases (added in the elementwise phase) ----
    const int cr = c, cz = 64 + c, cn = 128 + c;
    const float br0 = b_ih0[cr] + b_hh0[cr];
    const float bz0 = b_ih0[cz] + b_hh0[cz];
    const float bi0 = b_ih0[cn];
    const float bh0 = b_hh0[cn];
    const float br1 = b_ih1[cr] + b_hh1[cr];
    const float bz1 = b_ih1[cz] + b_hh1[cz];
    const float bi1 = b_ih1[cn];
    const float bh1 = b_hh1[cn];
    const float fw  = fc_w[c];

    // ---- zero-init hidden states ----
#pragma unroll
    for (int i = tid; i < 16 * 64; i += 128) { lh0[i] = (_Float16)0.f; lh1[i] = (_Float16)0.f; }
    const v8f zc = splat8(0.f);   // persistent zero C operand for first WMMA of each chain
    v8f h0 = zc, h1 = zc;
    __syncthreads();

    // x A-fragment source: lane ln -> batch row b0+ln; K chunk select by half-wave
    const float* xrow = x + ((size_t)(b0 + ln) * Tt) * Dd + (hl << 3);

    for (int t = 0; t < Tt; ++t) {
        // ---- phase A: read previous states + stream x ----
        v16h ah0a = ldA(lh0, 0, ln, hl), ah0b = ldA(lh0, 32, ln, hl);
        v16h ah1a = ldA(lh1, 0, ln, hl), ah1b = ldA(lh1, 32, ln, hl);

        v4f a0 = *(const v4f*)(xrow);
        v4f a1 = *(const v4f*)(xrow + 4);
        v4f a2 = *(const v4f*)(xrow + 16);
        v4f a3 = *(const v4f*)(xrow + 20);
        __builtin_prefetch(xrow + Dd, 0, 0);   // next timestep (global_prefetch_b8)
        v16h ax;
        ax[0]=(_Float16)a0[0]; ax[1]=(_Float16)a0[1]; ax[2]=(_Float16)a0[2]; ax[3]=(_Float16)a0[3];
        ax[4]=(_Float16)a1[0]; ax[5]=(_Float16)a1[1]; ax[6]=(_Float16)a1[2]; ax[7]=(_Float16)a1[3];
        ax[8]=(_Float16)a2[0]; ax[9]=(_Float16)a2[1]; ax[10]=(_Float16)a2[2]; ax[11]=(_Float16)a2[3];
        ax[12]=(_Float16)a3[0]; ax[13]=(_Float16)a3[1]; ax[14]=(_Float16)a3[2]; ax[15]=(_Float16)a3[3];
        xrow += Dd;

        __syncthreads();   // B1: all state reads done -> safe to overwrite lh0/lh1

        // ---- layer 0: 9 WMMAs (zero C, bias folded into elementwise) ----
        v8f ar = WMMA16(ax, *(const v16h*)(wf + 0 * 128 * 16), zc);
        ar = WMMA16(ah0a, Bhh0_r0, ar);
        ar = WMMA16(ah0b, Bhh0_r1, ar);
        v8f az = WMMA16(ax, *(const v16h*)(wf + 1 * 128 * 16), zc);
        az = WMMA16(ah0a, Bhh0_z0, az);
        az = WMMA16(ah0b, Bhh0_z1, az);
        v8f ai = WMMA16(ax, *(const v16h*)(wf + 2 * 128 * 16), zc);
        v8f ah = WMMA16(ah0a, Bhh0_n0, zc);
        ah = WMMA16(ah0b, Bhh0_n1, ah);
#pragma unroll
        for (int i = 0; i < 8; ++i) {
            float r = sig_f(ar[i] + br0);
            float z = sig_f(az[i] + bz0);
            float n = tanh_f(ai[i] + bi0 + r * (ah[i] + bh0));
            h0[i] = n + z * (h0[i] - n);
        }
#pragma unroll
        for (int i = 0; i < 8; ++i)
            lh0[(i + 8 * hl) * 64 + c] = (_Float16)h0[i];

        __syncthreads();   // B2: h0[t] visible to all waves

        // ---- layer 1: input is h0[t] (12 WMMAs) ----
        v16h an0 = ldA(lh0, 0, ln, hl), an1 = ldA(lh0, 32, ln, hl);

        v8f cr1 = WMMA16(an0,  *(const v16h*)(wf + 3 * 128 * 16), zc);
        cr1 = WMMA16(an1,  *(const v16h*)(wf + 4 * 128 * 16), cr1);
        cr1 = WMMA16(ah1a, Bhh1_r0, cr1);
        cr1 = WMMA16(ah1b, Bhh1_r1, cr1);
        v8f cz1 = WMMA16(an0,  *(const v16h*)(wf + 5 * 128 * 16), zc);
        cz1 = WMMA16(an1,  *(const v16h*)(wf + 6 * 128 * 16), cz1);
        cz1 = WMMA16(ah1a, Bhh1_z0, cz1);
        cz1 = WMMA16(ah1b, Bhh1_z1, cz1);
        v8f ci1 = WMMA16(an0,  *(const v16h*)(wf + 7 * 128 * 16), zc);
        ci1 = WMMA16(an1,  *(const v16h*)(wf + 8 * 128 * 16), ci1);
        v8f ch1 = WMMA16(ah1a, Bhh1_n0, zc);
        ch1 = WMMA16(ah1b, Bhh1_n1, ch1);
#pragma unroll
        for (int i = 0; i < 8; ++i) {
            float r = sig_f(cr1[i] + br1);
            float z = sig_f(cz1[i] + bz1);
            float n = tanh_f(ci1[i] + bi1 + r * (ch1[i] + bh1));
            h1[i] = n + z * (h1[i] - n);
        }
#pragma unroll
        for (int i = 0; i < 8; ++i)
            lh1[(i + 8 * hl) * 64 + c] = (_Float16)h1[i];

        __syncthreads();   // B3: h1[t] complete before next step's reads
    }

    // ---- head: out[b] = h1_last[b,:] . fc_w + fc_b ----
#pragma unroll
    for (int i = 0; i < 8; ++i)
        red[(i + 8 * hl) * 64 + c] = h1[i] * fw;
    __syncthreads();
    if (tid < 16) {
        float s = fc_b[0];
#pragma unroll
        for (int cc = 0; cc < 64; ++cc) s += red[tid * 64 + cc];
        out[b0 + tid] = s;
    }
}

// ---------------- launcher ----------------

extern "C" void kernel_launch(void* const* d_in, const int* in_sizes, int n_in,
                              void* d_out, int out_size, void* d_ws, size_t ws_size,
                              hipStream_t stream) {
    (void)in_sizes; (void)n_in; (void)out_size; (void)ws_size;

    const float* x     = (const float*)d_in[0];
    const float* W_ih0 = (const float*)d_in[1];
    const float* W_hh0 = (const float*)d_in[2];
    const float* b_ih0 = (const float*)d_in[3];
    const float* b_hh0 = (const float*)d_in[4];
    const float* W_ih1 = (const float*)d_in[5];
    const float* W_hh1 = (const float*)d_in[6];
    const float* b_ih1 = (const float*)d_in[7];
    const float* b_hh1 = (const float*)d_in[8];
    const float* fc_w  = (const float*)d_in[9];
    const float* fc_b  = (const float*)d_in[10];

    // f16 weight mirrors in workspace (86 KB total), 32B-aligned offsets
    _Float16* ws   = (_Float16*)d_ws;
    _Float16* wih0 = ws;              // 192*32  = 6144
    _Float16* whh0 = ws + 6144;       // 192*64  = 12288
    _Float16* wih1 = ws + 18432;      // 12288
    _Float16* whh1 = ws + 30720;      // 12288

    cvt_f16_kernel<<<(6144  + 255) / 256, 256, 0, stream>>>(W_ih0, wih0, 6144);
    cvt_f16_kernel<<<(12288 + 255) / 256, 256, 0, stream>>>(W_hh0, whh0, 12288);
    cvt_f16_kernel<<<(12288 + 255) / 256, 256, 0, stream>>>(W_ih1, wih1, 12288);
    cvt_f16_kernel<<<(12288 + 255) / 256, 256, 0, stream>>>(W_hh1, whh1, 12288);

    gru2_fused_wmma<<<Bb / 16, 128, 0, stream>>>(
        x, wih0, whh0, wih1, whh1,
        b_ih0, b_hh0, b_ih1, b_hh1, fc_w, fc_b, (float*)d_out);
}
